// Transformer_5033701670963
// MI455X (gfx1250) — matrix operand order repaired and checked by reading, not compiled, not run on hardware
//
// CDNA5 / gfx1250 (MI455X) transformer forward.
//
// Strategy (compile-only, reasoned from MI455X specs):
//  * ~235 GFLOP of GEMMs dominate; fp32 weights (~200MB) cost only ~9us of
//    HBM traffic at 23.3 TB/s, so we convert weights to f16 once per launch
//    and run every matmul on v_wmma_f32_16x16x32_f16 (f32 accumulate).
//  * wave32 / WGP: 256-thread blocks = 8 waves; each block owns a 128x128 C
//    tile, each wave 32x64 (2x4 WMMA accumulators), K stepped by 32 through
//    LDS (b128 staged, padded rows).
//  * Attention is run per (batch,head): scores = Q.K^T (NT), rel = Q.E^T (NT),
//    fused scale+skew+mask+softmax -> f16 probabilities, out = P.V (NN).
//  * Fragment layouts per ISA 7.12.2: A lane<->M + K across VGPRs;
//    B lane<->N + K across VGPRs (row-major [K,N] loads are natural);
//    C lane<->N, VGPR<->M.

#include <hip/hip_runtime.h>

typedef _Float16 h16;
typedef __attribute__((ext_vector_type(16))) _Float16 v16h;
typedef __attribute__((ext_vector_type(8)))  float    v8f;

#define B_   2
#define L_   1024
#define D_   512
#define H_   8
#define DP_  64
#define DFF_ 2048
#define V_   32000
#define BT_  (B_ * L_)          // 2048 token rows

#define BM 128
#define BN 128
#define BK 32
#define PADA 8
#define PADB 8

struct __attribute__((aligned(16))) U128 { unsigned x, y, z, w; };

union FragHU { unsigned u[8]; v16h v; };

// ---------------------------------------------------------------------------
// GEMM: C[M,N](f32,+bias,+relu, optional f16 mirror) = A[M,K]f16 * B f16
// bt==0: B row-major [K,N] (natural WMMA B layout).
// bt==1: B row-major [N,K] (i.e. C = A * B^T), staged transposed via LDS.
// ---------------------------------------------------------------------------
__global__ __launch_bounds__(256) void gemm_wmma_f16(
    const h16* __restrict__ A, const h16* __restrict__ Bm,
    float* __restrict__ C, h16* __restrict__ Ch,
    const float* __restrict__ bias,
    int M, int N, int K, int relu, int bt)
{
  __shared__ h16 As[BM][BK + PADA];
  __shared__ h16 Bs[BK][BN + PADB];

  const int tid  = threadIdx.x;
  const int lane = tid & 31;
  const int wave = tid >> 5;
  const int wm   = wave & 3;           // 4 waves along M
  const int wn   = wave >> 2;          // 2 waves along N
  const int bm0  = blockIdx.y * BM;
  const int bn0  = blockIdx.x * BN;

  v8f acc[2][4];
  #pragma unroll
  for (int i = 0; i < 2; ++i)
    #pragma unroll
    for (int j = 0; j < 4; ++j)
      #pragma unroll
      for (int r = 0; r < 8; ++r) acc[i][j][r] = 0.0f;

  const int arow    = lane & 15;             // A fragment: lane -> M row
  const int ahalf   = (lane >> 4) * 8;       // K half select
  const int bk_lane = (lane & 15) + ((lane >> 4) << 4); // unused; equals lane

  for (int k0 = 0; k0 < K; k0 += BK) {
    // ---- stage A tile: BM x BK halves (b128 per chunk) ----
    #pragma unroll
    for (int it = 0; it < 2; ++it) {
      int c  = tid + it * 256;               // 512 chunks of 8 halves
      int r  = c >> 2;
      int cc = (c & 3) * 8;
      U128 val = {0u, 0u, 0u, 0u};
      int gr = bm0 + r;
      if (gr < M) val = *(const U128*)(A + (size_t)gr * K + k0 + cc);
      *(U128*)&As[r][cc] = val;
    }
    // ---- stage B tile: BK x BN halves ----
    if (bt == 0) {
      #pragma unroll
      for (int it = 0; it < 2; ++it) {
        int c  = tid + it * 256;             // 32 rows * 16 chunks
        int r  = c >> 4;
        int cc = (c & 15) * 8;
        U128 val = {0u, 0u, 0u, 0u};
        int gn = bn0 + cc;
        if (gn < N) val = *(const U128*)(Bm + (size_t)(k0 + r) * N + gn);
        *(U128*)&Bs[r][cc] = val;
      }
    } else {
      #pragma unroll
      for (int it = 0; it < 2; ++it) {
        int c  = tid + it * 256;             // 128 rows * 4 chunks
        int r  = c >> 2;                     // N index within tile
        int cc = (c & 3) * 8;                // K offset
        h16 tmp[8];
        U128 val = {0u, 0u, 0u, 0u};
        int gn = bn0 + r;
        if (gn < N) val = *(const U128*)(Bm + (size_t)gn * K + k0 + cc);
        *(U128*)tmp = val;
        #pragma unroll
        for (int e = 0; e < 8; ++e) Bs[cc + e][r] = tmp[e];
      }
    }
    __syncthreads();

    if (k0 + BK < K) {   // hint next A tile into cache (global_prefetch_b8)
      __builtin_prefetch(A + (size_t)(bm0 + (tid >> 1)) * K + k0 + BK, 0, 3);
    }

    FragHU afr[2], bfr[4];
    #pragma unroll
    for (int i = 0; i < 2; ++i) {
      int row = wm * 32 + i * 16 + arow;
      #pragma unroll
      for (int v = 0; v < 8; ++v) {
        int kk = ahalf + 2 * (v & 3) + ((v >> 2) << 4);
        afr[i].u[v] = *(const unsigned*)&As[row][kk];
      }
    }
    #pragma unroll
    for (int j = 0; j < 4; ++j) {
      int cb = wn * 64 + j * 16;
      #pragma unroll
      for (int v = 0; v < 8; ++v) {
        int kk = ahalf + 2 * (v & 3) + ((v >> 2) << 4);
        bfr[j].v[2 * v]     = Bs[kk][cb + arow];
        bfr[j].v[2 * v + 1] = Bs[kk + 1][cb + arow];
      }
    }
    #pragma unroll
    for (int i = 0; i < 2; ++i)
      #pragma unroll
      for (int j = 0; j < 4; ++j)
        acc[i][j] = __builtin_amdgcn_wmma_f32_16x16x32_f16(
            false, afr[i].v, false, bfr[j].v, (short)0, acc[i][j], false, false);
    __syncthreads();
  }

  // ---- epilogue: C lane<->N, VGPR r<->M (+8 for lanes 16..31) ----
  #pragma unroll
  for (int i = 0; i < 2; ++i) {
    int m0 = bm0 + wm * 32 + i * 16 + ((lane >> 4) << 3);
    #pragma unroll
    for (int j = 0; j < 4; ++j) {
      int n = bn0 + wn * 64 + j * 16 + (lane & 15);
      if (n < N) {
        float bv = bias ? bias[n] : 0.0f;
        #pragma unroll
        for (int r = 0; r < 8; ++r) {
          int m = m0 + r;
          if (m < M) {
            float v = acc[i][j][r] + bv;
            if (relu) v = fmaxf(v, 0.0f);
            C[(size_t)m * N + n] = v;
            if (Ch) Ch[(size_t)m * N + n] = (h16)v;
          }
        }
      }
    }
  }
}

// ---------------------------------------------------------------------------
// Elementwise kernels
// ---------------------------------------------------------------------------
__global__ void wconv_kernel(const float* __restrict__ src, h16* __restrict__ dst, int n) {
  int i = blockIdx.x * 256 + threadIdx.x;
  if (i < n) dst[i] = (h16)src[i];
}

// embedding gather * sqrt(D) + positional encoding; writes f32 + f16
__global__ void embed_kernel(const int* __restrict__ tok, const float* __restrict__ emb,
                             float* __restrict__ xf, h16* __restrict__ xh) {
  int i   = blockIdx.x * 256 + threadIdx.x;   // exactly B_*L_*D_ threads
  int d   = i & (D_ - 1);
  int t   = i >> 9;                           // token row (b*L + pos)
  int pos = t & (L_ - 1);
  float v  = emb[(size_t)tok[t] * D_ + d] * 22.62741699796952f;  // sqrt(512)
  int   i2 = d & ~1;
  float ang = (float)pos * __powf(10000.0f, -(float)i2 * (1.0f / (float)D_));
  v += (d & 1) ? __cosf(ang) : __sinf(ang);
  xf[i] = v;
  xh[i] = (h16)v;
}

// f32 [B*L, 512] -> f16 [B, H, L, 64]
__global__ void pack_heads_kernel(const float* __restrict__ src, h16* __restrict__ dst) {
  int i   = blockIdx.x * 256 + threadIdx.x;
  int d   = i & (D_ - 1);
  int t   = i >> 9;
  int b   = t >> 10;
  int pos = t & (L_ - 1);
  int h   = d >> 6;
  int dd  = d & 63;
  dst[(((size_t)(b * H_ + h) * L_) + pos) * DP_ + dd] = (h16)src[i];
}

// f32 [B, H, L, 64] -> f16 [B*L, 512]
__global__ void unpack_heads_kernel(const float* __restrict__ src, h16* __restrict__ dst) {
  int i   = blockIdx.x * 256 + threadIdx.x;
  int d   = i & (D_ - 1);
  int t   = i >> 9;
  int b   = t >> 10;
  int pos = t & (L_ - 1);
  int h   = d >> 6;
  int dd  = d & 63;
  dst[i] = (h16)src[(((size_t)(b * H_ + h) * L_) + pos) * DP_ + dd];
}

// scores: s = (raw + srel)/sqrt(64) + NEG*mask ; row softmax -> f16 probs.
// srel[q,k] = (k<=q) ? QE[q, k-q+L-1] : 0 (analytic form of the _skew trick).
__global__ __launch_bounds__(256) void softmax_rel_kernel(
    const float* __restrict__ sc, const float* __restrict__ qe,
    h16* __restrict__ wout, const int* __restrict__ tok, int causal)
{
  __shared__ float red[8];
  const int q   = blockIdx.x;
  const int tid = threadIdx.x;
  float v[4];
  float mx = -3.4e38f;
  #pragma unroll
  for (int i = 0; i < 4; ++i) {
    int k = tid + i * 256;
    float s = sc[(size_t)q * L_ + k];
    if (k <= q) s += qe[(size_t)q * L_ + (k - q + L_ - 1)];
    s *= 0.125f;                               // 1/sqrt(DEPTH)
    if (tok[k] == 0 || (causal && k > q)) s += -1.0e9f;
    v[i] = s;
    mx = fmaxf(mx, s);
  }
  for (int o = 16; o > 0; o >>= 1) mx = fmaxf(mx, __shfl_xor(mx, o, 32));
  if ((tid & 31) == 0) red[tid >> 5] = mx;
  __syncthreads();
  mx = red[0];
  #pragma unroll
  for (int i = 1; i < 8; ++i) mx = fmaxf(mx, red[i]);
  __syncthreads();
  float sum = 0.0f;
  #pragma unroll
  for (int i = 0; i < 4; ++i) { v[i] = __expf(v[i] - mx); sum += v[i]; }
  for (int o = 16; o > 0; o >>= 1) sum += __shfl_xor(sum, o, 32);
  if ((tid & 31) == 0) red[tid >> 5] = sum;
  __syncthreads();
  sum = 0.0f;
  #pragma unroll
  for (int i = 0; i < 8; ++i) sum += red[i];
  float inv = 1.0f / sum;
  #pragma unroll
  for (int i = 0; i < 4; ++i)
    wout[(size_t)q * L_ + tid + i * 256] = (h16)(v[i] * inv);
}

// out = LayerNorm(x + r) ; writes f32 (may alias x) and f16 mirror
__global__ __launch_bounds__(256) void ln_res_kernel(
    const float* __restrict__ x, const float* __restrict__ r,
    const float* __restrict__ g, const float* __restrict__ b,
    float* __restrict__ of, h16* __restrict__ oh)
{
  __shared__ float red[8];
  const int row = blockIdx.x;
  const int tid = threadIdx.x;
  const size_t base = (size_t)row * D_;
  float a0 = x[base + tid]       + r[base + tid];
  float a1 = x[base + tid + 256] + r[base + tid + 256];
  float s = a0 + a1;
  for (int o = 16; o > 0; o >>= 1) s += __shfl_xor(s, o, 32);
  if ((tid & 31) == 0) red[tid >> 5] = s;
  __syncthreads();
  s = 0.0f;
  #pragma unroll
  for (int i = 0; i < 8; ++i) s += red[i];
  float mu = s * (1.0f / (float)D_);
  __syncthreads();
  float d0 = a0 - mu, d1 = a1 - mu;
  float s2 = d0 * d0 + d1 * d1;
  for (int o = 16; o > 0; o >>= 1) s2 += __shfl_xor(s2, o, 32);
  if ((tid & 31) == 0) red[tid >> 5] = s2;
  __syncthreads();
  s2 = 0.0f;
  #pragma unroll
  for (int i = 0; i < 8; ++i) s2 += red[i];
  float inv = rsqrtf(s2 * (1.0f / (float)D_) + 1e-6f);
  float o0 = g[tid] * d0 * inv + b[tid];
  float o1 = g[tid + 256] * d1 * inv + b[tid + 256];
  of[base + tid] = o0;       of[base + tid + 256] = o1;
  oh[base + tid] = (h16)o0;  oh[base + tid + 256] = (h16)o1;
}

// ---------------------------------------------------------------------------
// Host orchestration.
// Input order assumption: jax tree leaves (dicts alphabetical, lists in order):
//   0 inp, 1 tar, 2 dec_emb,
//   3 + 28*l .. : dec layer l = { ffn{b1,b2,w1,w2}, ln1{b,g}, ln2{b,g},
//                                 ln3{b,g}, mha1{E,bk,bo,bq,bv,wk,wo,wq,wv},
//                                 mha2{E,bk,bo,bq,bv,wk,wo,wq,wv} }
//   115 enc_emb,
//   116 + 17*l ..: enc layer l = { ffn{b1,b2,w1,w2}, ln1{b,g}, ln2{b,g},
//                                  mha{E,bk,bo,bq,bv,wk,wo,wq,wv} }
//   184 final_b, 185 final_w
// ---------------------------------------------------------------------------
extern "C" void kernel_launch(void* const* d_in, const int* in_sizes, int n_in,
                              void* d_out, int out_size, void* d_ws, size_t ws_size,
                              hipStream_t stream) {
  (void)in_sizes; (void)out_size; (void)ws_size;
  if (n_in < 186) return;

  const int* inp = (const int*)d_in[0];
  const int* tar = (const int*)d_in[1];
  auto P = [&](int i) -> const float* { return (const float*)d_in[i]; };

  // bump allocator over workspace (256B aligned)
  size_t off = 0;
  auto alloc = [&](size_t bytes) -> char* {
    char* p = (char*)d_ws + off;
    off += (bytes + 255) & ~(size_t)255;
    return p;
  };

  // f32 -> f16 weight conversion (deterministic; re-done every launch)
  auto cv = [&](int idx, size_t n) -> h16* {
    h16* p = (h16*)alloc(n * sizeof(h16));
    wconv_kernel<<<(int)((n + 255) / 256), 256, 0, stream>>>(P(idx), p, (int)n);
    return p;
  };

  struct MhaW { h16 *wq, *wk, *wv, *wo, *E; const float *bq, *bk, *bv, *bo; };
  struct FfnW { h16 *w1, *w2; const float *b1, *b2; };

  // ---- decoder params ----
  MhaW decA1[4], decA2[4];
  FfnW decF[4];
  const float* decLn[4][6];  // g1,b1,g2,b2,g3,b3
  for (int l = 0; l < 4; ++l) {
    int e = 3 + 28 * l;
    decF[l].b1 = P(e + 0); decF[l].b2 = P(e + 1);
    decF[l].w1 = cv(e + 2, (size_t)D_ * DFF_);
    decF[l].w2 = cv(e + 3, (size_t)DFF_ * D_);
    decLn[l][1] = P(e + 4); decLn[l][0] = P(e + 5);
    decLn[l][3] = P(e + 6); decLn[l][2] = P(e + 7);
    decLn[l][5] = P(e + 8); decLn[l][4] = P(e + 9);
    MhaW* A[2] = { &decA1[l], &decA2[l] };
    for (int a = 0; a < 2; ++a) {
      int m = e + 10 + 9 * a;
      A[a]->E  = cv(m + 0, (size_t)L_ * DP_);
      A[a]->bk = P(m + 1); A[a]->bo = P(m + 2);
      A[a]->bq = P(m + 3); A[a]->bv = P(m + 4);
      A[a]->wk = cv(m + 5, (size_t)D_ * D_);
      A[a]->wo = cv(m + 6, (size_t)D_ * D_);
      A[a]->wq = cv(m + 7, (size_t)D_ * D_);
      A[a]->wv = cv(m + 8, (size_t)D_ * D_);
    }
  }
  // ---- encoder params ----
  MhaW encA[4];
  FfnW encF[4];
  const float* encLn[4][4];  // g1,b1,g2,b2
  for (int l = 0; l < 4; ++l) {
    int e = 116 + 17 * l;
    encF[l].b1 = P(e + 0); encF[l].b2 = P(e + 1);
    encF[l].w1 = cv(e + 2, (size_t)D_ * DFF_);
    encF[l].w2 = cv(e + 3, (size_t)DFF_ * D_);
    encLn[l][1] = P(e + 4); encLn[l][0] = P(e + 5);
    encLn[l][3] = P(e + 6); encLn[l][2] = P(e + 7);
    encA[l].E  = cv(e + 8, (size_t)L_ * DP_);
    encA[l].bk = P(e + 9);  encA[l].bo = P(e + 10);
    encA[l].bq = P(e + 11); encA[l].bv = P(e + 12);
    encA[l].wk = cv(e + 13, (size_t)D_ * D_);
    encA[l].wo = cv(e + 14, (size_t)D_ * D_);
    encA[l].wq = cv(e + 15, (size_t)D_ * D_);
    encA[l].wv = cv(e + 16, (size_t)D_ * D_);
  }
  h16* final16 = cv(185, (size_t)D_ * V_);
  const float* final_b = P(184);

  // ---- activation workspace ----
  float* x_f  = (float*)alloc((size_t)BT_ * D_ * 4);    // encoder stream / enc_out
  h16*   x_h  = (h16*)  alloc((size_t)BT_ * D_ * 2);
  float* y_f  = (float*)alloc((size_t)BT_ * D_ * 4);    // decoder stream
  h16*   y_h  = (h16*)  alloc((size_t)BT_ * D_ * 2);
  float* t0   = (float*)alloc((size_t)BT_ * D_ * 4);
  float* t1   = (float*)alloc((size_t)BT_ * D_ * 4);
  float* t2   = (float*)alloc((size_t)BT_ * D_ * 4);
  h16*   cat_h= (h16*)  alloc((size_t)BT_ * D_ * 2);
  h16*   Qh   = (h16*)  alloc((size_t)B_ * H_ * L_ * DP_ * 2);
  h16*   Kh   = (h16*)  alloc((size_t)B_ * H_ * L_ * DP_ * 2);
  h16*   Vh   = (h16*)  alloc((size_t)B_ * H_ * L_ * DP_ * 2);
  float* Oh   = (float*)alloc((size_t)B_ * H_ * L_ * DP_ * 4);
  float* sc   = (float*)alloc((size_t)L_ * L_ * 4);
  float* qebuf= (float*)alloc((size_t)L_ * L_ * 4);
  h16*   wsm  = (h16*)  alloc((size_t)L_ * L_ * 2);
  float* hb   = (float*)alloc((size_t)BT_ * DFF_ * 4);
  h16*   hb_h = (h16*)  alloc((size_t)BT_ * DFF_ * 2);

  const int EW = (BT_ * D_) / 256;   // 4096 blocks for elementwise over tokens

  auto GEMM = [&](const h16* A, const h16* Bmx, float* C, h16* Ch,
                  const float* bias, int M, int N, int K, int relu, int bt) {
    dim3 grid((N + BN - 1) / BN, (M + BM - 1) / BM);
    gemm_wmma_f16<<<grid, 256, 0, stream>>>(A, Bmx, C, Ch, bias, M, N, K, relu, bt);
  };

  auto run_attn = [&](const MhaW& w, const h16* q_in, const h16* kv_in,
                      const int* mask_tok, int causal, float* attn_out) {
    GEMM(q_in,  w.wq, t0, nullptr, w.bq, BT_, D_, D_, 0, 0);
    GEMM(kv_in, w.wk, t1, nullptr, w.bk, BT_, D_, D_, 0, 0);
    GEMM(kv_in, w.wv, t2, nullptr, w.bv, BT_, D_, D_, 0, 0);
    pack_heads_kernel<<<EW, 256, 0, stream>>>(t0, Qh);
    pack_heads_kernel<<<EW, 256, 0, stream>>>(t1, Kh);
    pack_heads_kernel<<<EW, 256, 0, stream>>>(t2, Vh);
    for (int b = 0; b < B_; ++b) {
      for (int h = 0; h < H_; ++h) {
        size_t o = ((size_t)b * H_ + h) * L_ * DP_;
        GEMM(Qh + o, Kh + o, sc,    nullptr, nullptr, L_, L_, DP_, 0, 1); // Q.K^T
        GEMM(Qh + o, w.E,   qebuf,  nullptr, nullptr, L_, L_, DP_, 0, 1); // Q.E^T
        softmax_rel_kernel<<<L_, 256, 0, stream>>>(sc, qebuf, wsm,
                                                   mask_tok + b * L_, causal);
        GEMM(wsm, Vh + o, Oh + o, nullptr, nullptr, L_, DP_, L_, 0, 0);   // P.V
      }
    }
    unpack_heads_kernel<<<EW, 256, 0, stream>>>(Oh, cat_h);
    GEMM(cat_h, w.wo, attn_out, nullptr, w.bo, BT_, D_, D_, 0, 0);
  };

  // ================= encoder =================
  embed_kernel<<<EW, 256, 0, stream>>>(inp, P(115), x_f, x_h);
  for (int l = 0; l < 4; ++l) {
    run_attn(encA[l], x_h, x_h, inp, 0, t0);
    ln_res_kernel<<<BT_, 256, 0, stream>>>(x_f, t0, encLn[l][0], encLn[l][1], x_f, x_h);
    GEMM(x_h,  encF[l].w1, hb, hb_h, encF[l].b1, BT_, DFF_, D_, 1, 0);
    GEMM(hb_h, encF[l].w2, t0, nullptr, encF[l].b2, BT_, D_, DFF_, 0, 0);
    ln_res_kernel<<<BT_, 256, 0, stream>>>(x_f, t0, encLn[l][2], encLn[l][3], x_f, x_h);
  }
  // x_f / x_h now hold enc_out

  // ================= decoder =================
  embed_kernel<<<EW, 256, 0, stream>>>(tar, P(2), y_f, y_h);
  for (int l = 0; l < 4; ++l) {
    run_attn(decA1[l], y_h, y_h, tar, 1, t0);                // causal self-attn
    ln_res_kernel<<<BT_, 256, 0, stream>>>(y_f, t0, decLn[l][0], decLn[l][1], y_f, y_h);
    run_attn(decA2[l], y_h, x_h, inp, 0, t0);                // cross-attn
    ln_res_kernel<<<BT_, 256, 0, stream>>>(y_f, t0, decLn[l][2], decLn[l][3], y_f, y_h);
    GEMM(y_h,  decF[l].w1, hb, hb_h, decF[l].b1, BT_, DFF_, D_, 1, 0);
    GEMM(hb_h, decF[l].w2, t0, nullptr, decF[l].b2, BT_, D_, DFF_, 0, 0);
    ln_res_kernel<<<BT_, 256, 0, stream>>>(y_f, t0, decLn[l][4], decLn[l][5], y_f, y_h);
  }

  // ================= final logits =================
  GEMM(y_h, final16, (float*)d_out, nullptr, final_b, BT_, V_, D_, 0, 0);
}
